// MLA_10144712753496
// MI455X (gfx1250) — compile-verified, matrix operand order checked by reading, not run on hardware
//
#include <hip/hip_runtime.h>

typedef __attribute__((ext_vector_type(16))) _Float16 v16h;
typedef __attribute__((ext_vector_type(8)))  _Float16 v8h;
typedef __attribute__((ext_vector_type(8)))  float    v8f;

#define KPAD 40   // halves per LDS row (32 data + 8 pad), keeps 16B alignment

// ---------------------------------------------------------------------------
// Fused 1x1-conv GEMM: Out[b] = relu( (A @ B[b]) * scale[m] + bias[m] )
//   A: M x K (weights fp32 row-major), B: K x N per batch (fp32, N contiguous)
// Block tile 64(M) x 128(N), K-step 32, 8 waves (2x4), wave tile 32x32.
// Double-buffered LDS; f16 WMMA fragments, fp32 accumulation.
// Staging: A row-major (v8h stores); B gathered per-column so every LDS write
// is a contiguous ds_store_b128 and every global load coalesces across lanes.
// ---------------------------------------------------------------------------
__global__ __launch_bounds__(256)
void gemm_bn_relu(const float* __restrict__ A,
                  const float* __restrict__ Bmat,
                  const float* __restrict__ scale,
                  const float* __restrict__ bias,
                  float* __restrict__ Out,
                  int M, int N, int K, long strideB, long strideO)
{
    __shared__ _Float16 lds_a[2][64 * KPAD];    // 64 M-rows x 32 K
    __shared__ _Float16 lds_b[2][128 * KPAD];   // 128 N-cols x 32 K (K-contig)

    const int tid    = threadIdx.x;
    const int lane   = tid & 31;
    const int wave   = tid >> 5;
    const int wm     = wave & 1;        // 2 waves along M
    const int wn     = wave >> 1;       // 4 waves along N
    const int l16    = lane & 15;
    const int halfid = lane >> 4;
    const int kbase  = halfid << 3;     // K offset of this half's 8-half run

    const int b  = blockIdx.z;
    const int m0 = blockIdx.y << 6;
    const int n0 = blockIdx.x << 7;

    const float* __restrict__ Bp = Bmat + (long)b * strideB;
    float* __restrict__       Op = Out  + (long)b * strideO;

    // staging roles (fixed per thread)
    const int arow = tid >> 2;          // 0..63
    const int ac0  = (tid & 3) << 3;    // 0,8,16,24
    const int bn   = tid & 127;         // owned B column within tile
    const int kg0  = tid >> 7;          // 0..1 ; thread covers kg0 and kg0+2

    v8f acc[2][2] = {};
    float4 a0, a1;
    float  rb[16];

    const int nsteps = K >> 5;

    // ---- prologue: load + stage tile 0 ----
    {
        const float4* ga = (const float4*)(A + (size_t)(m0 + arow) * K + ac0);
        a0 = ga[0]; a1 = ga[1];
        #pragma unroll
        for (int r = 0; r < 2; r++) {
            const int kg = kg0 + 2 * r;
            #pragma unroll
            for (int j = 0; j < 8; j++)
                rb[r * 8 + j] = Bp[(size_t)(kg * 8 + j) * N + n0 + bn];
        }
        v8h ha;
        ha[0] = (_Float16)a0.x; ha[1] = (_Float16)a0.y;
        ha[2] = (_Float16)a0.z; ha[3] = (_Float16)a0.w;
        ha[4] = (_Float16)a1.x; ha[5] = (_Float16)a1.y;
        ha[6] = (_Float16)a1.z; ha[7] = (_Float16)a1.w;
        *(v8h*)&lds_a[0][arow * KPAD + ac0] = ha;
        #pragma unroll
        for (int r = 0; r < 2; r++) {
            v8h hb;
            #pragma unroll
            for (int j = 0; j < 8; j++) hb[j] = (_Float16)rb[r * 8 + j];
            *(v8h*)&lds_b[0][bn * KPAD + (kg0 + 2 * r) * 8] = hb;
        }
    }
    __syncthreads();

    for (int t = 0; t < nsteps; t++) {
        const int  cur  = t & 1;
        const bool more = (t + 1) < nsteps;

        // ---- issue next tile's global loads early (latency hidden by WMMA) --
        if (more) {
            const int k0 = (t + 1) << 5;
            const float4* ga = (const float4*)(A + (size_t)(m0 + arow) * K + k0 + ac0);
            a0 = ga[0]; a1 = ga[1];
            #pragma unroll
            for (int r = 0; r < 2; r++) {
                const int kg = kg0 + 2 * r;
                #pragma unroll
                for (int j = 0; j < 8; j++)
                    rb[r * 8 + j] = Bp[(size_t)(k0 + kg * 8 + j) * N + n0 + bn];
            }
        }

        // ---- fragments from LDS[cur] (ISA 16-bit A/B layouts), 4 WMMAs ----
        v16h af[2], bfr[2];
        #pragma unroll
        for (int i = 0; i < 2; i++) {
            const int ar = (wm << 5) + (i << 4) + l16;
            v8h lo = *(const v8h*)&lds_a[cur][ar * KPAD + kbase];
            v8h hi = *(const v8h*)&lds_a[cur][ar * KPAD + 16 + kbase];
            af[i] = __builtin_shufflevector(lo, hi, 0,1,2,3,4,5,6,7,8,9,10,11,12,13,14,15);
        }
        #pragma unroll
        for (int j = 0; j < 2; j++) {
            const int br = (wn << 5) + (j << 4) + l16;
            v8h lo = *(const v8h*)&lds_b[cur][br * KPAD + kbase];
            v8h hi = *(const v8h*)&lds_b[cur][br * KPAD + 16 + kbase];
            bfr[j] = __builtin_shufflevector(lo, hi, 0,1,2,3,4,5,6,7,8,9,10,11,12,13,14,15);
        }
        #pragma unroll
        for (int i = 0; i < 2; i++)
            #pragma unroll
            for (int j = 0; j < 2; j++)
                acc[i][j] = __builtin_amdgcn_wmma_f32_16x16x32_f16(
                    false, af[i], false, bfr[j], (short)0, acc[i][j], false, false);

        // ---- convert + store next tile into LDS[1-cur] (disjoint buffer) ----
        if (more) {
            v8h ha;
            ha[0] = (_Float16)a0.x; ha[1] = (_Float16)a0.y;
            ha[2] = (_Float16)a0.z; ha[3] = (_Float16)a0.w;
            ha[4] = (_Float16)a1.x; ha[5] = (_Float16)a1.y;
            ha[6] = (_Float16)a1.z; ha[7] = (_Float16)a1.w;
            *(v8h*)&lds_a[1 - cur][arow * KPAD + ac0] = ha;
            #pragma unroll
            for (int r = 0; r < 2; r++) {
                v8h hb;
                #pragma unroll
                for (int j = 0; j < 8; j++) hb[j] = (_Float16)rb[r * 8 + j];
                *(v8h*)&lds_b[1 - cur][bn * KPAD + (kg0 + 2 * r) * 8] = hb;
            }
        }
        __syncthreads();
    }

    // ---- epilogue: BN scale/bias + ReLU; D layout: VGPR r -> M = r + 8*half --
    #pragma unroll
    for (int i = 0; i < 2; i++) {
        #pragma unroll
        for (int r = 0; r < 8; r++) {
            const int m  = m0 + (wm << 5) + (i << 4) + (halfid << 3) + r;
            const float sc = scale[m];
            const float bi = bias[m];
            #pragma unroll
            for (int j = 0; j < 2; j++) {
                const int n = n0 + (wn << 5) + (j << 4) + l16;
                float v = fmaf(acc[i][j][r], sc, bi);
                Op[(size_t)m * N + n] = fmaxf(v, 0.0f);
            }
        }
    }
}

// ---------------------------------------------------------------------------
// Depthwise 5x5 (pad 2) + per-channel scale/bias pair:
//   out = pw_w[c] * (dw5x5(in)[c] + dw_b[c]) + pw_b[c]
// One block per (channel, batch); whole 64x64 channel + halo staged in LDS.
// ---------------------------------------------------------------------------
__global__ __launch_bounds__(256)
void dwconv5x5(const float* __restrict__ in,
               const float* __restrict__ wdw, const float* __restrict__ bdw,
               const float* __restrict__ wpw, const float* __restrict__ bpw,
               float* __restrict__ out, int C)
{
    __shared__ float tile[68 * 68];
    const int c = blockIdx.x;
    const int b = blockIdx.y;
    const float* __restrict__ src = in  + ((size_t)b * C + c) * 4096;
    float* __restrict__       dst = out + ((size_t)b * C + c) * 4096;

    for (int i = threadIdx.x; i < 68 * 68; i += 256) {
        const int y = (i / 68) - 2;
        const int x = (i % 68) - 2;
        float v = 0.0f;
        if ((unsigned)x < 64u && (unsigned)y < 64u) v = src[y * 64 + x];
        tile[i] = v;
    }
    float w[25];
    #pragma unroll
    for (int i = 0; i < 25; i++) w[i] = wdw[c * 25 + i];   // uniform -> SGPR
    const float bd = bdw[c], pw = wpw[c], pb = bpw[c];
    __syncthreads();

    for (int p = threadIdx.x; p < 4096; p += 256) {
        const int y = p >> 6, x = p & 63;
        float s = 0.0f;
        #pragma unroll
        for (int ky = 0; ky < 5; ky++)
            #pragma unroll
            for (int kx = 0; kx < 5; kx++)
                s = fmaf(tile[(y + ky) * 68 + (x + kx)], w[ky * 5 + kx], s);
        dst[p] = fmaf(pw, s + bd, pb);
    }
}

// ---------------------------------------------------------------------------
// ReLU linear attention, one block per (head, batch). d = 8, N = 4096.
// Phase 1: vk[8][8] = sum_n v_d * relu(k_e),  ksum[8] = sum_n relu(k_e)
//          (register accumulators, wave shfl_xor + LDS cross-wave reduce)
// Phase 2: out[d][n] = (vk @ relu(q))_d / (ksum . relu(q) + eps)
// ---------------------------------------------------------------------------
__global__ __launch_bounds__(256)
void linattn(const float* __restrict__ qkv, const float* __restrict__ agg,
             float* __restrict__ out)
{
    const int h = blockIdx.x;      // 0..63
    const int b = blockIdx.y;      // 0..15
    const float* __restrict__ buf = (h < 32) ? qkv : agg;
    const int hh = (h < 32) ? h : h - 32;
    const float* __restrict__ qp = buf + ((size_t)b * 768 + hh * 24 + 0)  * 4096;
    const float* __restrict__ kp = buf + ((size_t)b * 768 + hh * 24 + 8)  * 4096;
    const float* __restrict__ vp = buf + ((size_t)b * 768 + hh * 24 + 16) * 4096;

    const int lane = threadIdx.x & 31;
    const int wave = threadIdx.x >> 5;

    float acc[72];                 // [0..63] = vk[d*8+e], [64..71] = ksum[e]
    #pragma unroll
    for (int t = 0; t < 72; t++) acc[t] = 0.0f;

    for (int n = threadIdx.x; n < 4096; n += 256) {
        float kr[8], vr[8];
        #pragma unroll
        for (int e = 0; e < 8; e++) kr[e] = fmaxf(kp[(size_t)e * 4096 + n], 0.0f);
        #pragma unroll
        for (int d = 0; d < 8; d++) vr[d] = vp[(size_t)d * 4096 + n];
        #pragma unroll
        for (int d = 0; d < 8; d++)
            #pragma unroll
            for (int e = 0; e < 8; e++)
                acc[d * 8 + e] = fmaf(vr[d], kr[e], acc[d * 8 + e]);
        #pragma unroll
        for (int e = 0; e < 8; e++) acc[64 + e] += kr[e];
    }

    __shared__ float red[8][72];
    __shared__ float fin[72];
    #pragma unroll
    for (int t = 0; t < 72; t++) {
        float s = acc[t];
        #pragma unroll
        for (int off = 16; off > 0; off >>= 1) s += __shfl_xor(s, off, 32);
        if (lane == 0) red[wave][t] = s;
    }
    __syncthreads();
    if (threadIdx.x < 72) {
        float s = 0.0f;
        #pragma unroll
        for (int w = 0; w < 8; w++) s += red[w][threadIdx.x];
        fin[threadIdx.x] = s;
    }
    __syncthreads();

    float vkr[64], ksr[8];
    #pragma unroll
    for (int t = 0; t < 64; t++) vkr[t] = fin[t];
    #pragma unroll
    for (int e = 0; e < 8; e++)  ksr[e] = fin[64 + e];

    float* __restrict__ op = out + ((size_t)b * 512 + h * 8) * 4096;
    for (int n = threadIdx.x; n < 4096; n += 256) {
        float qe[8];
        #pragma unroll
        for (int e = 0; e < 8; e++) qe[e] = fmaxf(qp[(size_t)e * 4096 + n], 0.0f);
        float denom = 1e-5f;
        #pragma unroll
        for (int e = 0; e < 8; e++) denom = fmaf(ksr[e], qe[e], denom);
        const float inv = 1.0f / denom;
        #pragma unroll
        for (int d = 0; d < 8; d++) {
            float num = 0.0f;
            #pragma unroll
            for (int e = 0; e < 8; e++) num = fmaf(vkr[d * 8 + e], qe[e], num);
            op[(size_t)d * 4096 + n] = num * inv;
        }
    }
}

// ---------------------------------------------------------------------------
extern "C" void kernel_launch(void* const* d_in, const int* in_sizes, int n_in,
                              void* d_out, int out_size, void* d_ws, size_t ws_size,
                              hipStream_t stream)
{
    const float* x      = (const float*)d_in[0];
    const float* qkv_w  = (const float*)d_in[1];
    const float* qkv_s  = (const float*)d_in[2];
    const float* qkv_b  = (const float*)d_in[3];
    const float* dw_w   = (const float*)d_in[4];
    const float* dw_b   = (const float*)d_in[5];
    const float* pw_w   = (const float*)d_in[6];
    const float* pw_b   = (const float*)d_in[7];
    const float* proj_w = (const float*)d_in[8];
    const float* proj_s = (const float*)d_in[9];
    const float* proj_b = (const float*)d_in[10];
    float* out = (float*)d_out;

    float* qkvbuf  = (float*)d_ws;                       // 16x768x4096 f32 (201 MB)
    float* aggbuf  = qkvbuf + (size_t)16 * 768 * 4096;   // 16x768x4096 f32 (201 MB)
    float* attnbuf = aggbuf + (size_t)16 * 768 * 4096;   // 16x512x4096 f32 (134 MB)

    // qkv = relu(bn(W_qkv @ x)) : M=768, K=256, N=4096 per batch
    gemm_bn_relu<<<dim3(32, 12, 16), 256, 0, stream>>>(
        qkv_w, x, qkv_s, qkv_b, qkvbuf,
        768, 4096, 256, 256L * 4096, 768L * 4096);

    // agg = pw*(dw5x5(qkv)+bd)+pb
    dwconv5x5<<<dim3(768, 16), 256, 0, stream>>>(
        qkvbuf, dw_w, dw_b, pw_w, pw_b, aggbuf, 768);

    // ReLU linear attention over ms = concat(qkv, agg)
    linattn<<<dim3(64, 16), 256, 0, stream>>>(qkvbuf, aggbuf, attnbuf);

    // y = relu(bn(W_proj @ attn)) : M=256, K=512, N=4096 per batch
    gemm_bn_relu<<<dim3(32, 4, 16), 256, 0, stream>>>(
        proj_w, attnbuf, proj_s, proj_b, out,
        256, 4096, 512, 512L * 4096, 256L * 4096);
}
